// AttentionHAN_77747497992626
// MI455X (gfx1250) — matrix operand order repaired and checked by reading, not compiled, not run on hardware
//
#include <hip/hip_runtime.h>
#include <hip/hip_bf16.h>

typedef __attribute__((ext_vector_type(16))) _Float16 v16h;
typedef __attribute__((ext_vector_type(8)))  float    v8f;
typedef __attribute__((ext_vector_type(4)))  int      v4i_;

#define B_ROWS 131072
#define K_IN   256

// Async global->LDS staging (gfx1250 GLOBAL_LOAD_ASYNC_TO_LDS_B128), guarded
// so we fall back to the proven VGPR-bounce copy if the toolchain lacks it.
#if defined(__has_builtin)
# if __has_builtin(__builtin_amdgcn_global_load_async_to_lds_b128) && \
     __has_builtin(__builtin_amdgcn_s_wait_asynccnt)
#  define USE_ASYNC_LDS 1
# endif
#endif
#ifndef USE_ASYNC_LDS
# define USE_ASYNC_LDS 0
#endif

#if USE_ASYNC_LDS
typedef __attribute__((address_space(1))) v4i_* gv4i_ptr;
typedef __attribute__((address_space(3))) v4i_* lv4i_ptr;
#endif

// ---------------------------------------------------------------------------
// Kernel 0: prep — f16 weight stacks, stacked biases, zero chi counts
// ---------------------------------------------------------------------------
__global__ __launch_bounds__(256) void prep_kernel(
    const float* __restrict__ Wtq, const float* __restrict__ Wtv,
    const float* __restrict__ Wiq, const float* __restrict__ Wik,
    const float* __restrict__ Wiv,
    const float* __restrict__ btq, const float* __restrict__ btv,
    const float* __restrict__ biq, const float* __restrict__ bik,
    const float* __restrict__ biv,
    _Float16* __restrict__ W16T, _Float16* __restrict__ W16I,
    float* __restrict__ biasT, float* __restrict__ biasI,
    int* __restrict__ counts /* 1024 ints: [t 128x4][i 128x4] */)
{
    const int tid = blockIdx.x * blockDim.x + threadIdx.x;
    const int nth = gridDim.x * blockDim.x;

    // stacked text weights: rows 0-127 = Wtq, 128-255 = Wtv  (each 128x256)
    for (int i = tid; i < 2 * 32768; i += nth)
        W16T[i] = (_Float16)(i < 32768 ? Wtq[i] : Wtv[i - 32768]);

    // stacked image weights: Wiq | Wik | Wiv
    for (int i = tid; i < 3 * 32768; i += nth)
        W16I[i] = (_Float16)(i < 32768 ? Wiq[i]
                   : (i < 65536 ? Wik[i - 32768] : Wiv[i - 65536]));

    if (tid < 256) biasT[tid] = (tid < 128) ? btq[tid] : btv[tid - 128];
    if (tid < 384) biasI[tid] = (tid < 128) ? biq[tid]
                       : (tid < 256 ? bik[tid - 128] : biv[tid - 256]);
    if (tid < 1024) counts[tid] = 0;
}

// ---------------------------------------------------------------------------
// Kernel 1: fused GEMM  out[B x N] = A[B x 256] * W16^T + bias, plus chi counts
// on the last 128 columns (the V features).  N = 256 (text) or 384 (image).
// Block: 256 threads = 8 waves; M-tile = 32 rows; each wave owns one 16-row
// tile and N/64 column tiles of 16.
// ---------------------------------------------------------------------------
template<int N>
__global__ __launch_bounds__(256) void gemm_qkv(
    const float*    __restrict__ A,
    const _Float16* __restrict__ W16,   // [N x 256] f16, row-major
    const float*    __restrict__ bias,  // [N]
    const int*      __restrict__ label, // [B]
    float*          __restrict__ out,   // [B x N]
    int*            __restrict__ counts /* [128 x 4] */)
{
    constexpr int CTW    = (N / 16) / 4;  // column tiles per wave (4 or 6)
    constexpr int VSTART = N - 128;       // first "V" column

    __shared__ _Float16 Bs[N * 64];       // K-chunk of 64 for all N cols
    __shared__ int cnt[128 * 4];

    const int tid  = threadIdx.x;
    const int wave = tid >> 5;
    const int lane = tid & 31;
    const int l15  = lane & 15;
    const int lhi  = lane >> 4;           // 0 or 1

    for (int i = tid; i < 512; i += 256) cnt[i] = 0;

    const int m_base = blockIdx.x * 32 + (wave >> 2) * 16;
    const int ct0    = (wave & 3) * CTW;

    // ---- A fragments for this wave's 16 rows, full K=256, in registers ----
    // A 16x32 layout (contiguous, matches B): lanes 0-15 row M=l, K0..15;
    // lanes 16-31 same rows, K16..31.
    v16h afrag[8];
    {
        const float* Arow = A + (size_t)(m_base + l15) * K_IN;
        const int koff = lhi * 16;
        #pragma unroll
        for (int k0 = 0; k0 < 8; ++k0) {
            const int base = k0 * 32 + koff;
            const float4 x0 = *reinterpret_cast<const float4*>(Arow + base);
            const float4 x1 = *reinterpret_cast<const float4*>(Arow + base + 4);
            const float4 y0 = *reinterpret_cast<const float4*>(Arow + base + 8);
            const float4 y1 = *reinterpret_cast<const float4*>(Arow + base + 12);
            v16h f;
            f[0]  = (_Float16)x0.x; f[1]  = (_Float16)x0.y;
            f[2]  = (_Float16)x0.z; f[3]  = (_Float16)x0.w;
            f[4]  = (_Float16)x1.x; f[5]  = (_Float16)x1.y;
            f[6]  = (_Float16)x1.z; f[7]  = (_Float16)x1.w;
            f[8]  = (_Float16)y0.x; f[9]  = (_Float16)y0.y;
            f[10] = (_Float16)y0.z; f[11] = (_Float16)y0.w;
            f[12] = (_Float16)y1.x; f[13] = (_Float16)y1.y;
            f[14] = (_Float16)y1.z; f[15] = (_Float16)y1.w;
            afrag[k0] = f;
        }
    }

    v8f acc[CTW] = {};

    for (int ko = 0; ko < 4; ++ko) {
        __syncthreads();
        // stage W16[:, ko*64 .. +63] into LDS (16-byte chunks, uniform loop)
        const int kb = ko * 64;
#if USE_ASYNC_LDS
        for (int c = tid; c < N * 8; c += 256) {
            const int n  = c >> 3;
            const int kk = (c & 7) * 8;
            __builtin_amdgcn_global_load_async_to_lds_b128(
                (gv4i_ptr)&W16[n * K_IN + kb + kk],
                (lv4i_ptr)&Bs[n * 64 + kk],
                0, 0);
        }
        __builtin_amdgcn_s_wait_asynccnt(0);
#else
        for (int c = tid; c < N * 8; c += 256) {
            const int n  = c >> 3;
            const int kk = (c & 7) * 8;
            *reinterpret_cast<uint4*>(&Bs[n * 64 + kk]) =
                *reinterpret_cast<const uint4*>(&W16[n * K_IN + kb + kk]);
        }
#endif
        __syncthreads();

        #pragma unroll
        for (int ks = 0; ks < 2; ++ks) {
            const v16h a = afrag[ko * 2 + ks];
            #pragma unroll
            for (int ct = 0; ct < CTW; ++ct) {
                // B 32x16 in this code's contiguous k order (same as A): lane = col; lanes0-15 K0..15,
                // lanes16-31 K16..31 (16 contiguous f16 per lane)
                const int col = (ct0 + ct) * 16 + l15;
                const v16h b  = *reinterpret_cast<const v16h*>(
                                    &Bs[col * 64 + ks * 32 + lhi * 16]);
                acc[ct] = __builtin_amdgcn_wmma_f32_16x16x32_f16(
                    false, a, false, b, (short)0, acc[ct], false, false);
            }
        }
    }

    // ---- epilogue: bias, store, chi-square contingency counts ----
    int labs[8];
    #pragma unroll
    for (int r = 0; r < 8; ++r) labs[r] = label[m_base + lhi * 8 + r] & 1;

    #pragma unroll
    for (int ct = 0; ct < CTW; ++ct) {
        const int col   = (ct0 + ct) * 16 + l15;
        const float bcl = bias[col];
        #pragma unroll
        for (int r = 0; r < 8; ++r) {
            const int row = m_base + lhi * 8 + r;   // C layout: lanes16-31 = M+8
            const float v = acc[ct][r] + bcl;
            out[(size_t)row * N + col] = v;
            if (col >= VSTART) {
                const int fv = (v > 0.7f) ? 1 : 0;
                atomicAdd(&cnt[(col - VSTART) * 4 + fv * 2 + labs[r]], 1);
            }
        }
    }
    __syncthreads();
    for (int i = tid; i < 512; i += 256) {
        const int v = cnt[i];
        if (v) atomicAdd(&counts[i], v);
    }
}

// ---------------------------------------------------------------------------
// Kernel 2: chi-square per feature (faithful to reference's expected[a,b] =
// col[a]*row[b]/total quirk), global max, alphas[0..3]=t, [4..7]=i.
// One block, 256 threads: tid<128 -> t features, tid>=128 -> i features.
// ---------------------------------------------------------------------------
__global__ __launch_bounds__(256) void chi_alpha_kernel(
    const int* __restrict__ counts,  // [2][128][4]
    float* __restrict__ alphas)      // [8]
{
    __shared__ float red[256];
    const int tid = threadIdx.x;
    const int4 c4 = reinterpret_cast<const int4*>(counts)[tid];
    const float c00 = (float)c4.x, c01 = (float)c4.y;
    const float c10 = (float)c4.z, c11 = (float)c4.w;
    const float col0 = c00 + c10, col1 = c01 + c11;  // sum over f_val (per l)
    const float row0 = c00 + c01, row1 = c10 + c11;  // sum over l (per f_val)
    const float T = c00 + c01 + c10 + c11 + 1e-6f;
    const float e00 = col0 * row0 / T, e01 = col0 * row1 / T;
    const float e10 = col1 * row0 / T, e11 = col1 * row1 / T;
    float chi = 0.f;
    {
        float d;
        d = c00 - e00; chi += d * d / (e00 + 1e-6f);
        d = c01 - e01; chi += d * d / (e01 + 1e-6f);
        d = c10 - e10; chi += d * d / (e10 + 1e-6f);
        d = c11 - e11; chi += d * d / (e11 + 1e-6f);
    }
    red[tid] = chi;
    __syncthreads();
    for (int s = 128; s > 0; s >>= 1) {
        if (tid < s) red[tid] = fmaxf(red[tid], red[tid + s]);
        __syncthreads();
    }
    const float mx = red[0] + 1e-6f;
    if (tid < 4)                  alphas[tid]             = chi / mx;
    if (tid >= 128 && tid < 132)  alphas[4 + (tid - 128)] = chi / mx;
}

// ---------------------------------------------------------------------------
// Kernel 3: per-row attention score, fuse, 384->1 output GEMV.
// One wave per row; lane owns 4 consecutive columns of each 128-vector.
// ---------------------------------------------------------------------------
__global__ __launch_bounds__(256) void fuse_out_kernel(
    const float* __restrict__ outT,   // [B x 256]  (tQ | tV)
    const float* __restrict__ outI,   // [B x 384]  (iQ | iK | iV)
    const float* __restrict__ attn_W, // [4 x 64]
    const float* __restrict__ attn_b, // [4]
    const float* __restrict__ Wout,   // [1 x 384]
    const float* __restrict__ bout,   // [1]
    const float* __restrict__ alphas, // [8]
    float* __restrict__ y)            // [B]
{
    const int wave = threadIdx.x >> 5;
    const int lane = threadIdx.x & 31;
    const int h    = lane >> 3;        // head (32 cols per head, 4 cols/lane)
    const int c0   = lane * 4;
    const int d0   = c0 & 31;

    const float at = alphas[h], ai = alphas[4 + h];
    const float ab = attn_b[h], bo = bout[0];
    const float4 wq  = *reinterpret_cast<const float4*>(attn_W + h * 64 + d0);
    const float4 wk  = *reinterpret_cast<const float4*>(attn_W + h * 64 + 32 + d0);
    const float4 wf  = *reinterpret_cast<const float4*>(Wout + c0);
    const float4 wtq = *reinterpret_cast<const float4*>(Wout + 128 + c0);
    const float4 wiq = *reinterpret_cast<const float4*>(Wout + 256 + c0);

    for (int b = blockIdx.x * 8 + wave; b < B_ROWS; b += gridDim.x * 8) {
        const float* pT = outT + (size_t)b * 256;
        const float* pI = outI + (size_t)b * 384;
        const float4 tQ = *reinterpret_cast<const float4*>(pT + c0);
        const float4 tV = *reinterpret_cast<const float4*>(pT + 128 + c0);
        const float4 iQ = *reinterpret_cast<const float4*>(pI + c0);
        const float4 iK = *reinterpret_cast<const float4*>(pI + 128 + c0);
        const float4 iV = *reinterpret_cast<const float4*>(pI + 256 + c0);

        float s = tQ.x * wq.x + tQ.y * wq.y + tQ.z * wq.z + tQ.w * wq.w
                + iK.x * wk.x + iK.y * wk.y + iK.z * wk.z + iK.w * wk.w;
        s += __shfl_xor(s, 1, 32);
        s += __shfl_xor(s, 2, 32);
        s += __shfl_xor(s, 4, 32);   // all 8 lanes of the head now hold the sum

        const float score = 1.f / (1.f + __expf(-(s + ab)));
        const float e_t = score * at;
        const float e_i = score * ai;
        const float om  = 1.f - e_t;

        float4 fu;
        fu.x = tV.x * e_t + om * (iV.x * e_i);
        fu.y = tV.y * e_t + om * (iV.y * e_i);
        fu.z = tV.z * e_t + om * (iV.z * e_i);
        fu.w = tV.w * e_t + om * (iV.w * e_i);

        float a = fu.x * wf.x  + fu.y * wf.y  + fu.z * wf.z  + fu.w * wf.w
                + tQ.x * wtq.x + tQ.y * wtq.y + tQ.z * wtq.z + tQ.w * wtq.w
                + iQ.x * wiq.x + iQ.y * wiq.y + iQ.z * wiq.z + iQ.w * wiq.w;
        a += __shfl_xor(a, 1, 32);
        a += __shfl_xor(a, 2, 32);
        a += __shfl_xor(a, 4, 32);
        a += __shfl_xor(a, 8, 32);
        a += __shfl_xor(a, 16, 32);
        if (lane == 0) y[b] = a + bo;
    }
}

// ---------------------------------------------------------------------------
extern "C" void kernel_launch(void* const* d_in, const int* in_sizes, int n_in,
                              void* d_out, int out_size, void* d_ws, size_t ws_size,
                              hipStream_t stream) {
    (void)in_sizes; (void)n_in; (void)out_size; (void)ws_size;

    const float* text_vec  = (const float*)d_in[0];
    const float* image_vec = (const float*)d_in[1];
    const int*   label     = (const int*)d_in[2];
    const float* Wtq = (const float*)d_in[3];
    const float* btq = (const float*)d_in[4];
    // d_in[5], d_in[6] = Wtk, btk -- unused by the reference dataflow
    const float* Wtv = (const float*)d_in[7];
    const float* btv = (const float*)d_in[8];
    const float* Wiq = (const float*)d_in[9];
    const float* biq = (const float*)d_in[10];
    const float* Wik = (const float*)d_in[11];
    const float* bik = (const float*)d_in[12];
    const float* Wiv = (const float*)d_in[13];
    const float* biv = (const float*)d_in[14];
    const float* attn_W = (const float*)d_in[15];
    const float* attn_b = (const float*)d_in[16];
    const float* Wout   = (const float*)d_in[17];
    const float* bout   = (const float*)d_in[18];
    float* y = (float*)d_out;

    char* ws = (char*)d_ws;
    const size_t SZ_OUTT = (size_t)B_ROWS * 256 * sizeof(float);
    const size_t SZ_OUTI = (size_t)B_ROWS * 384 * sizeof(float);
    float*    outT   = (float*)(ws);
    float*    outI   = (float*)(ws + SZ_OUTT);
    _Float16* W16T   = (_Float16*)(ws + SZ_OUTT + SZ_OUTI);
    _Float16* W16I   = W16T + 256 * 256;
    float*    biasT  = (float*)(W16I + 384 * 256);
    float*    biasI  = biasT + 256;
    int*      counts = (int*)(biasI + 384);   // [t 512][i 512]
    float*    alphas = (float*)(counts + 1024);

    prep_kernel<<<256, 256, 0, stream>>>(Wtq, Wtv, Wiq, Wik, Wiv,
                                         btq, btv, biq, bik, biv,
                                         W16T, W16I, biasT, biasI, counts);

    gemm_qkv<256><<<B_ROWS / 32, 256, 0, stream>>>(
        text_vec, W16T, biasT, label, outT, counts);
    gemm_qkv<384><<<B_ROWS / 32, 256, 0, stream>>>(
        image_vec, W16I, biasI, label, outI, counts + 512);

    chi_alpha_kernel<<<1, 256, 0, stream>>>(counts, alphas);

    fuse_out_kernel<<<2048, 256, 0, stream>>>(
        outT, outI, attn_W, attn_b, Wout, bout, alphas, y);
}
